// HandwritingSeq2SeqModel_47742856462471
// MI455X (gfx1250) — matrix operand order repaired and checked by reading, not compiled, not run on hardware
//
#include <hip/hip_runtime.h>
#include <cstdint>
#include <cstddef>

// ---------------------------------------------------------------------------
// Model dims (fixed by the reference)
// ---------------------------------------------------------------------------
#define BB 64
#define TT 1024
#define UU 64
#define HH 256
#define VV 128

// ---------------------------------------------------------------------------
// WMMA types (CDNA5 gfx1250: V_WMMA_F32_16X16X32_BF16, wave32)
// ---------------------------------------------------------------------------
typedef __attribute__((ext_vector_type(16))) __bf16 v16bf;
typedef __attribute__((ext_vector_type(8)))  float  v8f;
typedef __attribute__((ext_vector_type(4)))  unsigned int uint32x4;
typedef __attribute__((ext_vector_type(8)))  int  int32x8;
typedef __attribute__((ext_vector_type(4)))  int  int32x4;

#if defined(__HIP_DEVICE_COMPILE__) && __has_builtin(__builtin_amdgcn_tensor_load_to_lds)
#define HAVE_TDM 1
#endif

__device__ __host__ __forceinline__ uint16_t f2bf(float f) {
  union { float f; uint32_t u; } v; v.f = f;
  uint32_t u = v.u;
  u += 0x7fffu + ((u >> 16) & 1u);   // round-to-nearest-even
  return (uint16_t)(u >> 16);
}
__device__ __host__ __forceinline__ float bf2f(uint16_t h) {
  union { uint32_t u; float f; } v; v.u = ((uint32_t)h) << 16;
  return v.f;
}

__device__ __forceinline__ v8f wmma_bf16(v16bf a, v16bf b, v8f c) {
#if defined(__HIP_DEVICE_COMPILE__)
  return __builtin_amdgcn_wmma_f32_16x16x32_bf16(false, a, false, b, (short)0, c,
                                                 false, false);
#else
  return c;
#endif
}

// A tile (16x32 bf16, MxK): lane L holds row m = L&15; K pattern per ISA:
//   VGPR j(0..3): K = (L>>4)*8 + 2j,2j+1 ; VGPR j(4..7): K = 16 + (L>>4)*8 + ...
__device__ __forceinline__ v16bf load_a_tile(const uint16_t* A, long lda,
                                             int m0, int k0) {
  v16bf a;
  uint16_t* ap = reinterpret_cast<uint16_t*>(&a);
  const int lane = threadIdx.x & 31;
  const uint16_t* r = A + (long)(m0 + (lane & 15)) * lda + k0 + ((lane >> 4) << 3);
#pragma unroll
  for (int j = 0; j < 4; ++j) { ap[2*j]   = r[2*j];      ap[2*j+1] = r[2*j+1]; }
#pragma unroll
  for (int j = 0; j < 4; ++j) { ap[8+2*j] = r[16+2*j];   ap[9+2*j] = r[16+2*j+1]; }
  return a;
}

// B tile (32x16 bf16, KxN): lane L holds column n0+(L&15); element e = K row, same K pattern as A.
__device__ __forceinline__ v16bf load_b_tile(const uint16_t* B, long ldb,
                                             int k0, int n0) {
  v16bf b;
  uint16_t* bp = reinterpret_cast<uint16_t*>(&b);
  const int lane = threadIdx.x & 31;
  const uint16_t* r = B + (long)(k0 + ((lane >> 4) << 3)) * ldb + n0 + (lane & 15);
#pragma unroll
  for (int e = 0; e < 16; ++e) bp[e] = r[(long)((e & 7) + ((e >> 3) << 4)) * ldb];
  return b;
}

// C/D tile (16x16 f32): lane L holds column n = L&15; VGPR r -> row r + (L>>4)*8.
__device__ __forceinline__ void store_c_f32(float* C, long ldc, int m0, int n0,
                                            v8f c, const float* bias) {
  const int lane = threadIdx.x & 31;
  const int n = n0 + (lane & 15);
  const float bv = bias ? bias[n] : 0.f;
  const int mb = m0 + ((lane >> 4) << 3);
#pragma unroll
  for (int r = 0; r < 8; ++r) C[(long)(mb + r) * ldc + n] = c[r] + bv;
}
__device__ __forceinline__ void store_c_bf16(uint16_t* C, long ldc, int m0, int n0,
                                             v8f c, const float* bias) {
  const int lane = threadIdx.x & 31;
  const int n = n0 + (lane & 15);
  const float bv = bias ? bias[n] : 0.f;
  const int mb = m0 + ((lane >> 4) << 3);
#pragma unroll
  for (int r = 0; r < 8; ++r) C[(long)(mb + r) * ldc + n] = f2bf(c[r] + bv);
}

// ---------------------------------------------------------------------------
// Generic WMMA GEMM: C[M,N](f32|bf16) = A[M,K]bf16 @ B[K,N]bf16 + bias
// One 16x16 C tile per wave. M%16==0, N%16==0, K%32==0.
// ---------------------------------------------------------------------------
template <bool BF16OUT>
__global__ void k_gemm(const uint16_t* __restrict__ A, const uint16_t* __restrict__ B,
                       void* __restrict__ C, const float* __restrict__ bias,
                       int M, int N, int K, long lda, long ldb, long ldc) {
  const int wave = (int)((blockIdx.x * blockDim.x + threadIdx.x) >> 5);
  const int ntn = N >> 4;
  const int total = (M >> 4) * ntn;
  if (wave >= total) return;
  const int m0 = (wave / ntn) << 4;
  const int n0 = (wave % ntn) << 4;
  v8f acc;
#pragma unroll
  for (int r = 0; r < 8; ++r) acc[r] = 0.f;
  const int lane = threadIdx.x & 31;
  for (int k0 = 0; k0 < K; k0 += 32) {
    if (k0 + 32 < K)  // hint next B rows into cache (global_prefetch_b8)
      __builtin_prefetch(B + (long)(k0 + 32 + lane) * ldb + n0, 0, 1);
    v16bf a = load_a_tile(A, lda, m0, k0);
    v16bf b = load_b_tile(B, ldb, k0, n0);
    acc = wmma_bf16(a, b, acc);
  }
  if (BF16OUT) store_c_bf16((uint16_t*)C, ldc, m0, n0, acc, bias);
  else         store_c_f32((float*)C, ldc, m0, n0, acc, bias);
}

// ---------------------------------------------------------------------------
// Bridge GEMM with TDM-staged B slab in LDS.
// Each block = 8 waves sharing one m-row block and 8 consecutive n-tiles
// (128 columns). The 512x128 bf16 B slab (128KB) is DMA'd into LDS by the
// Tensor Data Mover (tensor_load_to_lds + s_wait_tensorcnt), then all waves
// read B via LDS. Requires (N/16) % 8 == 0.
// ---------------------------------------------------------------------------
__global__ __launch_bounds__(256) void k_gemm_bstage(
    const uint16_t* __restrict__ A, const uint16_t* __restrict__ B,
    uint16_t* __restrict__ C, const float* __restrict__ bias,
    int M, int N, int K, long lda, long ldb, long ldc) {
  extern __shared__ uint16_t bsh[];   // [K][128] bf16
  const int ntn  = N >> 4;
  const int first = blockIdx.x * 8;           // first wave tile in this block
  const int m0   = (first / ntn) << 4;
  const int nb0  = (first % ntn) << 4;        // block column base (multiple of 128)
  const int wave = threadIdx.x >> 5;

#ifdef HAVE_TDM
  if (wave == 0) {
    // ---- Tensor DMA descriptor (D#), per CDNA5 ISA 8.3/8.4 ----
    const uint64_t gaddr = (uint64_t)(uintptr_t)(B + nb0);
    const uint32_t lds_off = (uint32_t)(uintptr_t)(&bsh[0]);
    uint32x4 g0;
    g0[0] = 1u;                                   // count=1, user descriptor
    g0[1] = lds_off;                              // lds_addr
    g0[2] = (uint32_t)gaddr;                      // global_addr[31:0]
    g0[3] = (uint32_t)((gaddr >> 32) & 0x01FFFFFFu) | (2u << 30);  // addr hi | type=2
    const uint32_t td0 = 128u, td1 = (uint32_t)K; // tensor dims (elements)
    const uint32_t t0  = 128u, t1  = (uint32_t)K; // tile dims
    const uint64_t s0  = (uint64_t)ldb;           // dim0 stride (elements)
    int32x8 g1;
    g1[0] = (int)(1u << 16);                                            // data_size=2B
    g1[1] = (int)((td0 & 0xFFFFu) << 16);                               // tensor_dim0 lo
    g1[2] = (int)(((td0 >> 16) & 0xFFFFu) | ((td1 & 0xFFFFu) << 16));   // td0 hi | td1 lo
    g1[3] = (int)(((td1 >> 16) & 0xFFFFu) | ((t0 & 0xFFFFu) << 16));    // td1 hi | tile0
    g1[4] = (int)(t1 & 0xFFFFu);                                        // tile1 | tile2=0
    g1[5] = (int)(uint32_t)(s0 & 0xFFFFFFFFu);                          // stride0 lo
    g1[6] = (int)(uint32_t)((s0 >> 32) & 0xFFFFu);                      // stride0 hi
    g1[7] = 0;
    int32x4 gz; gz[0] = 0; gz[1] = 0; gz[2] = 0; gz[3] = 0;             // groups 2/3 unused (2D)
#if defined(__clang_major__) && __clang_major__ >= 23
    int32x8 gz8;
    gz8[0] = 0; gz8[1] = 0; gz8[2] = 0; gz8[3] = 0;
    gz8[4] = 0; gz8[5] = 0; gz8[6] = 0; gz8[7] = 0;
    __builtin_amdgcn_tensor_load_to_lds(g0, g1, gz, gz, gz8, 0);
#else
    __builtin_amdgcn_tensor_load_to_lds(g0, g1, gz, gz, 0);
#endif
    __builtin_amdgcn_s_wait_tensorcnt(0);
  }
  __syncthreads();
#else
  for (int i = threadIdx.x; i < K * 128; i += 256)
    bsh[i] = B[(long)(i >> 7) * ldb + nb0 + (i & 127)];
  __syncthreads();
#endif

  const int n0loc = wave << 4;
  v8f acc;
#pragma unroll
  for (int r = 0; r < 8; ++r) acc[r] = 0.f;
  for (int k0 = 0; k0 < K; k0 += 32) {
    v16bf a = load_a_tile(A, lda, m0, k0);
    v16bf b = load_b_tile(bsh, 128, k0, n0loc);
    acc = wmma_bf16(a, b, acc);
  }
  store_c_bf16(C, ldc, m0, nb0 + n0loc, acc, bias);
}

// ---------------------------------------------------------------------------
// Weight prep: f32 -> bf16 (plain and transposed)
// ---------------------------------------------------------------------------
__global__ void k_cvt(const float* __restrict__ s, uint16_t* __restrict__ d, long n) {
  long i = (long)blockIdx.x * blockDim.x + threadIdx.x;
  if (i < n) d[i] = f2bf(s[i]);
}
// src [R][C] f32 row-major  ->  dst [C][R] bf16 row-major
__global__ void k_tcvt(const float* __restrict__ s, uint16_t* __restrict__ d,
                       int R, int C) {
  long i = (long)blockIdx.x * blockDim.x + threadIdx.x;
  if (i < (long)R * C) {
    int r = (int)(i / C), c = (int)(i % C);
    d[(long)c * R + r] = f2bf(s[i]);
  }
}

// ---------------------------------------------------------------------------
// BiLSTM scan. One persistent workgroup (1024 thr = 32 waves) per direction.
// LDS: h (bf16, 32KB) + gate buffer g (f32, 256KB) = 288KB < 320KB/WGP.
// The recurrent weight slice of each wave (Whh cols [32w,32w+32), 16KB bf16 =
// 128 VGPRs) is preloaded ONCE into registers; the per-step critical path is
// then only ds_load(A=h) -> wmma -> LDS gate exchange -> VALU update.
// Layer0 input dim is 3 -> handled with 3 scalar FMAs in the elementwise phase.
// ---------------------------------------------------------------------------
struct LstmArgs {
  const float*    src;       // layer0 input  [B,T,3]
  const uint16_t* xin;       // layer1 input  [B,T,512] bf16
  const uint16_t* WhhT[2];   // [256][1024] bf16
  const uint16_t* WihT[2];   // layer1: [512][1024] bf16
  const float*    Wih0[2];   // layer0: [1024][3] f32
  const float*    bih[2];
  const float*    bhh[2];
  uint16_t*       xout;      // [B,T,512] bf16, dir writes columns dir*256..
};

template <bool L1>
__global__ __launch_bounds__(1024) void k_lstm_scan(LstmArgs p) {
  extern __shared__ uint8_t smem[];
  uint16_t* h_sh = (uint16_t*)smem;                  // [64][256] bf16
  float*    g_sh = (float*)(smem + BB * HH * 2);     // [64][1024] f32

  const int dir  = blockIdx.x;
  const int tid  = threadIdx.x;
  const int wave = tid >> 5;
  const int lane = tid & 31;
  const int n0   = wave << 5;          // gate columns owned by this wave

  for (int i = tid; i < BB * HH; i += 1024) h_sh[i] = 0;
  float c_reg[16];
#pragma unroll
  for (int j = 0; j < 16; ++j) c_reg[j] = 0.f;

  const int bown  = tid >> 4;          // batch row owned in elementwise phase
  const int hbase = (tid & 15) << 4;   // 16 hidden columns owned
  const uint16_t* WhhT = p.WhhT[dir];
  const uint16_t* WihT = L1 ? p.WihT[dir] : nullptr;
  const float* bih = p.bih[dir];
  const float* bhh = p.bhh[dir];

  // Preload the recurrent weight slice into VGPRs (lives across all 1024 steps)
  v16bf breg[HH / 32][2];
#pragma unroll
  for (int ks = 0; ks < HH / 32; ++ks) {
    breg[ks][0] = load_b_tile(WhhT, 1024, ks * 32, n0);
    breg[ks][1] = load_b_tile(WhhT, 1024, ks * 32, n0 + 16);
  }
  __syncthreads();

  for (int t = 0; t < TT; ++t) {
    const int tt = dir ? (TT - 1 - t) : t;
    // ---- GEMM phase ----
    v8f acc[4][2];
#pragma unroll
    for (int mt = 0; mt < 4; ++mt)
#pragma unroll
      for (int nt = 0; nt < 2; ++nt)
#pragma unroll
        for (int r = 0; r < 8; ++r) acc[mt][nt][r] = 0.f;

    if (L1) {  // input projection: A rows = batch, K over 512 (prev layer out)
      const uint16_t* abase = p.xin + (long)tt * 512;
      for (int k0 = 0; k0 < 512; k0 += 32) {
        v16bf b0 = load_b_tile(WihT, 1024, k0, n0);
        v16bf b1 = load_b_tile(WihT, 1024, k0, n0 + 16);
#pragma unroll
        for (int mt = 0; mt < 4; ++mt) {
          v16bf a = load_a_tile(abase, (long)TT * 512, mt * 16, k0);
          acc[mt][0] = wmma_bf16(a, b0, acc[mt][0]);
          acc[mt][1] = wmma_bf16(a, b1, acc[mt][1]);
        }
      }
    }
    // recurrent part: A = h from LDS, B = register-resident Whh slice
#pragma unroll
    for (int ks = 0; ks < HH / 32; ++ks) {
#pragma unroll
      for (int mt = 0; mt < 4; ++mt) {
        v16bf a = load_a_tile(h_sh, HH, mt * 16, ks * 32);
        acc[mt][0] = wmma_bf16(a, breg[ks][0], acc[mt][0]);
        acc[mt][1] = wmma_bf16(a, breg[ks][1], acc[mt][1]);
      }
    }
    {  // spill accumulators to the shared gate buffer
      const int n = lane & 15;
      const int mb = (lane >> 4) << 3;
#pragma unroll
      for (int mt = 0; mt < 4; ++mt)
#pragma unroll
        for (int r = 0; r < 8; ++r) {
          g_sh[(mt * 16 + mb + r) * 1024 + n0 + n]      = acc[mt][0][r];
          g_sh[(mt * 16 + mb + r) * 1024 + n0 + 16 + n] = acc[mt][1][r];
        }
    }
    __syncthreads();

    // ---- elementwise phase: thread owns h[bown][hbase..hbase+15] ----
    float x0 = 0.f, x1 = 0.f, x2 = 0.f;
    if (!L1) {
      const float* xr = p.src + ((long)bown * TT + tt) * 3;
      x0 = xr[0]; x1 = xr[1]; x2 = xr[2];
    }
#pragma unroll
    for (int j = 0; j < 16; ++j) {
      const int hh = hbase + j;
      float gi = g_sh[bown * 1024 + hh];
      float gf = g_sh[bown * 1024 + 256 + hh];
      float gg = g_sh[bown * 1024 + 512 + hh];
      float go = g_sh[bown * 1024 + 768 + hh];
      if (!L1) {
        const float* w = p.Wih0[dir];
        gi += w[(hh)       * 3] * x0 + w[(hh)       * 3 + 1] * x1 + w[(hh)       * 3 + 2] * x2;
        gf += w[(256 + hh) * 3] * x0 + w[(256 + hh) * 3 + 1] * x1 + w[(256 + hh) * 3 + 2] * x2;
        gg += w[(512 + hh) * 3] * x0 + w[(512 + hh) * 3 + 1] * x1 + w[(512 + hh) * 3 + 2] * x2;
        go += w[(768 + hh) * 3] * x0 + w[(768 + hh) * 3 + 1] * x1 + w[(768 + hh) * 3 + 2] * x2;
      }
      gi += bih[hh]       + bhh[hh];
      gf += bih[256 + hh] + bhh[256 + hh];
      gg += bih[512 + hh] + bhh[512 + hh];
      go += bih[768 + hh] + bhh[768 + hh];
      const float i_s = 1.f / (1.f + expf(-gi));
      const float f_s = 1.f / (1.f + expf(-gf));
      const float o_s = 1.f / (1.f + expf(-go));
      const float c = f_s * c_reg[j] + i_s * tanhf(gg);
      c_reg[j] = c;
      const float h = o_s * tanhf(c);
      const uint16_t hb = f2bf(h);
      h_sh[bown * HH + hh] = hb;
      p.xout[((long)bown * TT + tt) * 512 + dir * HH + hh] = hb;
    }
    __syncthreads();
  }
}

// ---------------------------------------------------------------------------
// Decoder helper kernels (one chain per decode step; all tiny)
// ---------------------------------------------------------------------------
__global__ void k_h0(const uint16_t* __restrict__ bridged, float* __restrict__ h,
                     uint16_t* __restrict__ hbf) {
  const int b = blockIdx.x, j = threadIdx.x;
  const float v = tanhf(bf2f(bridged[(long)b * TT * HH + j]));
  h[b * HH + j] = v;
  hbf[b * HH + j] = f2bf(v);
}

__global__ void k_embed(const int* __restrict__ tgt, int u,
                        const uint16_t* __restrict__ embed,
                        uint16_t* __restrict__ xin) {
  const int b = blockIdx.x, j = threadIdx.x;
  const int v = tgt[b * UU + u];
  xin[b * 768 + j] = embed[v * HH + j];
}

// attention: scores = h . bridged, masked softmax, ctx = attn @ enc_out
__global__ __launch_bounds__(256) void k_attn(
    const float* __restrict__ h, const uint16_t* __restrict__ bridged,
    const uint16_t* __restrict__ enc, const int* __restrict__ lens,
    uint16_t* __restrict__ xin_ctx, uint16_t* __restrict__ stepin_ctx) {
  __shared__ float hsh[HH];
  __shared__ float sc[TT];
  __shared__ float red[256];
  const int b = blockIdx.x, tid = threadIdx.x;
  hsh[tid] = h[b * HH + tid];
  __syncthreads();
  const int len = lens[b];
  for (int t = tid; t < TT; t += 256) {
    float s = 0.f;
    const uint16_t* br = bridged + ((long)b * TT + t) * HH;
    for (int k = 0; k < HH; ++k) s += hsh[k] * bf2f(br[k]);
    sc[t] = (t < len) ? s : -INFINITY;
  }
  __syncthreads();
  float m = -INFINITY;
  for (int t = tid; t < TT; t += 256) m = fmaxf(m, sc[t]);
  red[tid] = m;
  __syncthreads();
  for (int s = 128; s > 0; s >>= 1) {
    if (tid < s) red[tid] = fmaxf(red[tid], red[tid + s]);
    __syncthreads();
  }
  const float mx = red[0];
  __syncthreads();
  float sum = 0.f;
  for (int t = tid; t < TT; t += 256) {
    const float e = __expf(sc[t] - mx);
    sc[t] = e;
    sum += e;
  }
  red[tid] = sum;
  __syncthreads();
  for (int s = 128; s > 0; s >>= 1) {
    if (tid < s) red[tid] += red[tid + s];
    __syncthreads();
  }
  const float inv = 1.f / red[0];
  __syncthreads();
  for (int c = tid; c < 512; c += 256) {
    float acc = 0.f;
    const uint16_t* e0 = enc + (long)b * TT * 512 + c;
    for (int t = 0; t < TT; ++t) acc += sc[t] * bf2f(e0[(long)t * 512]);
    acc *= inv;
    const uint16_t cb = f2bf(acc);
    xin_ctx[b * 768 + 256 + c] = cb;
    stepin_ctx[b * 768 + 256 + c] = cb;
  }
}

__global__ void k_gru_update(const float* __restrict__ gi, const float* __restrict__ gh,
                             float* __restrict__ h, uint16_t* __restrict__ hbf,
                             uint16_t* __restrict__ stepin) {
  const int b = blockIdx.x, j = threadIdx.x;
  const float ir = gi[b * 768 + j], iz = gi[b * 768 + 256 + j], in = gi[b * 768 + 512 + j];
  const float hr = gh[b * 768 + j], hz = gh[b * 768 + 256 + j], hn = gh[b * 768 + 512 + j];
  const float r = 1.f / (1.f + expf(-(ir + hr)));
  const float z = 1.f / (1.f + expf(-(iz + hz)));
  const float n = tanhf(in + r * hn);
  const float hv = (1.f - z) * n + z * h[b * HH + j];
  h[b * HH + j] = hv;
  const uint16_t hb = f2bf(hv);
  hbf[b * HH + j] = hb;
  stepin[b * 768 + j] = hb;
}

__global__ void k_gelu(const float* __restrict__ x, uint16_t* __restrict__ y) {
  const int i = blockIdx.x * blockDim.x + threadIdx.x;
  const float v = x[i];
  y[i] = f2bf(0.5f * v * (1.f + erff(v * 0.70710678118654752f)));
}

// ---------------------------------------------------------------------------
// Host orchestration
// ---------------------------------------------------------------------------
extern "C" void kernel_launch(void* const* d_in, const int* in_sizes, int n_in,
                              void* d_out, int out_size, void* d_ws, size_t ws_size,
                              hipStream_t stream) {
  (void)in_sizes; (void)n_in; (void)out_size; (void)ws_size;
  const float* d_src  = (const float*)d_in[0];
  const int*   d_lens = (const int*)d_in[1];
  const int*   d_tgt  = (const int*)d_in[2];
  const float* eWih[2][2]; const float* eWhh[2][2];
  const float* ebih[2][2]; const float* ebhh[2][2];
  int idx = 3;
  for (int l = 0; l < 2; ++l)
    for (int d = 0; d < 2; ++d) {
      eWih[l][d] = (const float*)d_in[idx++];
      eWhh[l][d] = (const float*)d_in[idx++];
      ebih[l][d] = (const float*)d_in[idx++];
      ebhh[l][d] = (const float*)d_in[idx++];
    }
  const float* embed    = (const float*)d_in[idx++];
  const float* gru_Wih  = (const float*)d_in[idx++];
  const float* gru_Whh  = (const float*)d_in[idx++];
  const float* gru_bih  = (const float*)d_in[idx++];
  const float* gru_bhh  = (const float*)d_in[idx++];
  const float* bridge_W = (const float*)d_in[idx++];
  const float* bridge_b = (const float*)d_in[idx++];
  const float* out_W1   = (const float*)d_in[idx++];
  const float* out_b1   = (const float*)d_in[idx++];
  const float* out_W2   = (const float*)d_in[idx++];
  const float* out_b2   = (const float*)d_in[idx++];

  // workspace layout
  uint8_t* wp = (uint8_t*)d_ws;
  auto alloc = [&](size_t n) -> void* {
    void* r = wp; wp += (n + 255) & ~(size_t)255; return r;
  };
  uint16_t* WhhT[2][2];
  for (int l = 0; l < 2; ++l)
    for (int d = 0; d < 2; ++d) WhhT[l][d] = (uint16_t*)alloc((size_t)HH * 1024 * 2);
  uint16_t* WihT1[2];
  for (int d = 0; d < 2; ++d) WihT1[d] = (uint16_t*)alloc((size_t)512 * 1024 * 2);
  uint16_t* gruWihT  = (uint16_t*)alloc((size_t)768 * 768 * 2);
  uint16_t* gruWhhT  = (uint16_t*)alloc((size_t)256 * 768 * 2);
  uint16_t* bridgeWT = (uint16_t*)alloc((size_t)512 * 256 * 2);
  uint16_t* outW1T   = (uint16_t*)alloc((size_t)768 * 256 * 2);
  uint16_t* outW2T   = (uint16_t*)alloc((size_t)256 * 128 * 2);
  uint16_t* embedBF  = (uint16_t*)alloc((size_t)VV * HH * 2);
  uint16_t* x0buf    = (uint16_t*)alloc((size_t)BB * TT * 512 * 2);
  uint16_t* encbuf   = (uint16_t*)alloc((size_t)BB * TT * 512 * 2);
  uint16_t* bridged  = (uint16_t*)alloc((size_t)BB * TT * HH * 2);
  float*    d_h      = (float*)alloc((size_t)BB * HH * 4);
  uint16_t* d_hbf    = (uint16_t*)alloc((size_t)BB * HH * 2);
  uint16_t* d_xin    = (uint16_t*)alloc((size_t)BB * 768 * 2);
  uint16_t* d_stepin = (uint16_t*)alloc((size_t)BB * 768 * 2);
  float*    d_gi     = (float*)alloc((size_t)BB * 768 * 4);
  float*    d_gh     = (float*)alloc((size_t)BB * 768 * 4);
  float*    d_hid    = (float*)alloc((size_t)BB * HH * 4);
  uint16_t* d_hidbf  = (uint16_t*)alloc((size_t)BB * HH * 2);

  auto tcvt = [&](const float* s, uint16_t* d, int R, int C) {
    long n = (long)R * C;
    k_tcvt<<<dim3((unsigned)((n + 255) / 256)), dim3(256), 0, stream>>>(s, d, R, C);
  };
  for (int l = 0; l < 2; ++l)
    for (int d = 0; d < 2; ++d) tcvt(eWhh[l][d], WhhT[l][d], 1024, 256);
  for (int d = 0; d < 2; ++d) tcvt(eWih[1][d], WihT1[d], 1024, 512);
  tcvt(gru_Wih, gruWihT, 768, 768);
  tcvt(gru_Whh, gruWhhT, 768, 256);
  tcvt(bridge_W, bridgeWT, 256, 512);
  tcvt(out_W1, outW1T, 256, 768);
  tcvt(out_W2, outW2T, 128, 256);
  k_cvt<<<dim3((VV * HH) / 256), dim3(256), 0, stream>>>(embed, embedBF, (long)VV * HH);

  // ---- encoder: 2 BiLSTM layers (persistent scan kernels, 288KB LDS) ----
  const size_t smem = (size_t)BB * HH * 2 + (size_t)BB * 1024 * 4;
  {
    LstmArgs a{};
    a.src = d_src; a.xin = nullptr; a.xout = x0buf;
    for (int d = 0; d < 2; ++d) {
      a.WhhT[d] = WhhT[0][d]; a.WihT[d] = nullptr; a.Wih0[d] = eWih[0][d];
      a.bih[d] = ebih[0][d]; a.bhh[d] = ebhh[0][d];
    }
    k_lstm_scan<false><<<dim3(2), dim3(1024), smem, stream>>>(a);
  }
  {
    LstmArgs a{};
    a.src = nullptr; a.xin = x0buf; a.xout = encbuf;
    for (int d = 0; d < 2; ++d) {
      a.WhhT[d] = WhhT[1][d]; a.WihT[d] = WihT1[d]; a.Wih0[d] = nullptr;
      a.bih[d] = ebih[1][d]; a.bhh[d] = ebhh[1][d];
    }
    k_lstm_scan<true><<<dim3(2), dim3(1024), smem, stream>>>(a);
  }

  // ---- bridge: [B*T,512] @ [512,256] -> bridged bf16 (TDM-staged B slab) ----
  k_gemm_bstage<<<dim3(8192), dim3(256), (size_t)512 * 128 * 2, stream>>>(
      encbuf, bridgeWT, bridged, bridge_b, BB * TT, HH, 512, 512, 256, 256);
  k_h0<<<dim3(BB), dim3(HH), 0, stream>>>(bridged, d_h, d_hbf);

  // ---- decoder: 64 steps ----
  float* out = (float*)d_out;
  for (int u = 0; u < UU; ++u) {
    k_embed<<<dim3(BB), dim3(HH), 0, stream>>>(d_tgt, u, embedBF, d_xin);
    k_attn<<<dim3(BB), dim3(256), 0, stream>>>(d_h, bridged, encbuf, d_lens,
                                               d_xin, d_stepin);
    // gi = [emb|ctx] @ gru_Wih^T + bih
    k_gemm<false><<<dim3(24), dim3(256), 0, stream>>>(
        d_xin, gruWihT, d_gi, gru_bih, BB, 768, 768, 768, 768, 768);
    // gh = h @ gru_Whh^T + bhh
    k_gemm<false><<<dim3(24), dim3(256), 0, stream>>>(
        d_hbf, gruWhhT, d_gh, gru_bhh, BB, 768, HH, HH, 768, 768);
    k_gru_update<<<dim3(BB), dim3(HH), 0, stream>>>(d_gi, d_gh, d_h, d_hbf, d_stepin);
    // hid = gelu([h|ctx] @ out_W1^T + b1)
    k_gemm<false><<<dim3(8), dim3(256), 0, stream>>>(
        d_stepin, outW1T, d_hid, out_b1, BB, HH, 768, 768, 256, 256);
    k_gelu<<<dim3(BB), dim3(HH), 0, stream>>>(d_hid, d_hidbf);
    // logits -> d_out[:, u, :]   (ldc = U*V strides over batch rows)
    k_gemm<false><<<dim3(4), dim3(256), 0, stream>>>(
        d_hidbf, outW2T, out + (long)u * VV, out_b2, BB, VV, HH, HH, VV,
        (long)UU * VV);
  }
}
